// CausalSelfAttention_35132832481430
// MI455X (gfx1250) — matrix operand order repaired and checked by reading, not compiled, not run on hardware
//
#include <hip/hip_runtime.h>

// ---------------------------------------------------------------------------
// CDNA5 (gfx1250) causal self-attention with RoPE + KV cache.
// Matmuls via V_WMMA_F32_16X16X32_BF16 (fp32 accumulate); GEMM tiles staged
// with GLOBAL_LOAD_ASYNC_TO_LDS_B128 (ASYNCcnt) — both weights pre-transposed
// so A and B^T tiles are contiguous-row copies.
// Shapes fixed: B=1, T=1024, OFF=1024, C=2048, H=16, hs=128, T_full=2048.
// ---------------------------------------------------------------------------

typedef __attribute__((ext_vector_type(16))) __bf16 v16bf;
typedef __attribute__((ext_vector_type(8)))  float  v8f;

struct U4x2 { uint4 a, b; };         // 32 bytes == v16bf

static __device__ __forceinline__ unsigned short f2bf(float f) {
  unsigned int u = __builtin_bit_cast(unsigned int, f);
  unsigned int r = u + 0x7FFFu + ((u >> 16) & 1u);   // round-to-nearest-even
  return (unsigned short)(r >> 16);
}

// Load 16 contiguous bf16 (32B, 16B-aligned halves) -> v16bf fragment.
static __device__ __forceinline__ v16bf load16(const unsigned short* p) {
  U4x2 t;
  t.a = *(const uint4*)p;
  t.b = *(const uint4*)(p + 8);
  return __builtin_bit_cast(v16bf, t);
}

// A-fragment (16-bit A 16x32): lane lr=row; regs hold K = lh*16+e
// -> one contiguous 16-element run from row pointer.
static __device__ __forceinline__ v16bf loadAfrag(const unsigned short* rowp, int lh) {
  U4x2 t;
  t.a = *(const uint4*)(rowp + lh * 16);
  t.b = *(const uint4*)(rowp + lh * 16 + 8);
  return __builtin_bit_cast(v16bf, t);
}

// Low 32 bits of a generic pointer to __shared__ == LDS byte offset.
static __device__ __forceinline__ unsigned lds_addr32(const void* p) {
  return (unsigned)(size_t)p;
}

// ---------------------------------------------------------------------------
// fp32 -> bf16 converter (straight copy)
// ---------------------------------------------------------------------------
__global__ __launch_bounds__(256) void cvt_kernel(const float* __restrict__ in,
                                                  unsigned short* __restrict__ out,
                                                  int n) {
  int i = blockIdx.x * 256 + threadIdx.x;
  if (i < n) out[i] = f2bf(in[i]);
}

// fp32 [K,N] -> bf16 [N,K] (transpose) so GEMM B tiles are contiguous rows.
__global__ __launch_bounds__(256) void cvt_transpose_kernel(
    const float* __restrict__ in, unsigned short* __restrict__ out,
    int K, int N) {
  int i = blockIdx.x * 256 + threadIdx.x;     // over K*N, coalesced read
  if (i < K * N) {
    int k = i / N, n = i - k * N;
    out[(size_t)n * K + k] = f2bf(in[i]);
  }
}

// ---------------------------------------------------------------------------
// bf16 GEMM: C[M,N](f32) = A[M,K](bf16) * BT[N,K](bf16)^T + bias[N]
// block tile 128x128, 8 waves (2x4 of 64x32 wave tiles), K-step 32.
// LDS tiles filled by GLOBAL_LOAD_ASYNC_TO_LDS_B128 (2 per thread per tile).
// ---------------------------------------------------------------------------
__global__ __launch_bounds__(256) void gemm_bf16_kernel(
    const unsigned short* __restrict__ A,   // [M,K]
    const unsigned short* __restrict__ BT,  // [N,K]
    const float* __restrict__ bias,
    float* __restrict__ C,
    int M, int N, int K) {
  __shared__ __align__(16) unsigned short As[128 * 32];  // [m][k]
  __shared__ __align__(16) unsigned short Bs[128 * 32];  // [n][k]

  const int tid  = threadIdx.x;
  const int lane = tid & 31;
  const int wave = tid >> 5;
  const int lr = lane & 15;
  const int lh = lane >> 4;
  const int m0 = blockIdx.y * 128;
  const int n0 = blockIdx.x * 128;
  const int wm = (wave >> 2) * 64;   // wave row base (0,64)
  const int wn = (wave & 3) * 32;    // wave col base (0..96)

  // staging assignment: each thread copies one 32B run of A and one of Bs
  const int srow = tid >> 1;             // 0..127
  const int sc0  = (tid & 1) * 16;       // 0 or 16 (elements)
  const unsigned ldsA = lds_addr32(&As[srow * 32 + sc0]);
  const unsigned ldsB = lds_addr32(&Bs[srow * 32 + sc0]);
  const unsigned aoff0 = (unsigned)(((size_t)(m0 + srow) * K + sc0) * 2);
  const unsigned boff0 = (unsigned)(((size_t)(n0 + srow) * K + sc0) * 2);

  v8f acc[4][2];
  v8f z = {};
#pragma unroll
  for (int mt = 0; mt < 4; ++mt)
#pragma unroll
    for (int nt = 0; nt < 2; ++nt) acc[mt][nt] = z;

  for (int k0 = 0; k0 < K; k0 += 32) {
    __syncthreads();  // LDS reuse: all waves done reading previous tile
    {
      unsigned aoff = aoff0 + (unsigned)(k0 * 2);
      unsigned boff = boff0 + (unsigned)(k0 * 2);
      asm volatile(
          "global_load_async_to_lds_b128 %0, %2, %4 offset:0\n\t"
          "global_load_async_to_lds_b128 %0, %2, %4 offset:16\n\t"
          "global_load_async_to_lds_b128 %1, %3, %5 offset:0\n\t"
          "global_load_async_to_lds_b128 %1, %3, %5 offset:16"
          :
          : "v"(ldsA), "v"(ldsB), "v"(aoff), "v"(boff), "s"(A), "s"(BT)
          : "memory");
      asm volatile("s_wait_asynccnt 0x0" ::: "memory");
    }
    __syncthreads();  // tile visible to all waves

    v16bf af[4], bf[2];
#pragma unroll
    for (int mt = 0; mt < 4; ++mt)
      af[mt] = loadAfrag(&As[(wm + mt * 16 + lr) * 32], lh);
#pragma unroll
    for (int nt = 0; nt < 2; ++nt)
      bf[nt] = load16(&Bs[(wn + nt * 16 + lr) * 32 + lh * 16]);  // K = lh*16+e

#pragma unroll
    for (int mt = 0; mt < 4; ++mt)
#pragma unroll
      for (int nt = 0; nt < 2; ++nt)
        acc[mt][nt] = __builtin_amdgcn_wmma_f32_16x16x32_bf16(
            false, af[mt], false, bf[nt], (short)0, acc[mt][nt], false, false);
  }

  // epilogue: C layout -> M = r + 8*lh, N = lr
#pragma unroll
  for (int mt = 0; mt < 4; ++mt)
#pragma unroll
    for (int nt = 0; nt < 2; ++nt) {
      int gcol = n0 + wn + nt * 16 + lr;
      float bv = bias[gcol];
#pragma unroll
      for (int r = 0; r < 8; ++r) {
        int grow = m0 + wm + mt * 16 + r + 8 * lh;
        C[(size_t)grow * N + gcol] = acc[mt][nt][r] + bv;
      }
    }
}

// ---------------------------------------------------------------------------
// RoPE new q/k, pass-through v; write fp32 k/v outputs + bf16 working copies
// qkv: [1024][6144]; qbf [h][t][128]; kbf [h][s][128]; vtbf [h][d][2048]
// kout/vout: [16][2048][128] fp32 (new tokens at s = 1024+t)
// ---------------------------------------------------------------------------
__global__ __launch_bounds__(256) void rope_concat_kernel(
    const float* __restrict__ qkv,
    unsigned short* __restrict__ qbf,
    unsigned short* __restrict__ kbf,
    unsigned short* __restrict__ vtbf,
    float* __restrict__ kout,
    float* __restrict__ vout) {
  const int t = blockIdx.x;
  const float pos = (float)(1024 + t);
  const float* row = qkv + (size_t)t * 6144;

#pragma unroll
  for (int i = 0; i < 4; ++i) {
    int ch = threadIdx.x + i * 256;   // 0..1023 half-pair index
    int h = ch >> 6, j = ch & 63;
    float inv_freq = __powf(10000.0f, -(float)j * (1.0f / 64.0f));
    float ang = pos * inv_freq;
    float cs, sn; __sincosf(ang, &sn, &cs);
    int base = h * 128 + j;
    // q
    float q1 = row[base], q2 = row[base + 64];
    size_t qo = (size_t)(h * 1024 + t) * 128 + j;
    qbf[qo]      = f2bf(q1 * cs - q2 * sn);
    qbf[qo + 64] = f2bf(q1 * sn + q2 * cs);
    // k
    float k1 = row[2048 + base], k2 = row[2048 + base + 64];
    float kr1 = k1 * cs - k2 * sn;
    float kr2 = k1 * sn + k2 * cs;
    size_t ko = (size_t)(h * 2048 + 1024 + t) * 128 + j;
    kout[ko] = kr1;        kout[ko + 64] = kr2;
    kbf[ko]  = f2bf(kr1);  kbf[ko + 64]  = f2bf(kr2);
  }
#pragma unroll
  for (int i = 0; i < 8; ++i) {
    int cfull = threadIdx.x * 8 + i;  // 0..2047
    int h = cfull >> 7, d = cfull & 127;
    float vv = row[4096 + cfull];
    vout[(size_t)(h * 2048 + 1024 + t) * 128 + d] = vv;
    vtbf[(size_t)(h * 128 + d) * 2048 + 1024 + t] = f2bf(vv);
  }
}

// ---------------------------------------------------------------------------
// Copy KV cache into outputs + bf16 working copies (s < 1024)
// ---------------------------------------------------------------------------
__global__ __launch_bounds__(256) void cache_concat_kernel(
    const float* __restrict__ ck, const float* __restrict__ cv,
    unsigned short* __restrict__ kbf, unsigned short* __restrict__ vtbf,
    float* __restrict__ kout, float* __restrict__ vout) {
  int i = blockIdx.x * 256 + threadIdx.x;   // < 16*1024*128
  int h = i >> 17;
  int rem = i & 131071;
  int s = rem >> 7, d = rem & 127;
  size_t o = (size_t)(h * 2048 + s) * 128 + d;
  float kc = ck[i];
  kout[o] = kc;
  kbf[o]  = f2bf(kc);
  float vc = cv[i];
  vout[o] = vc;
  vtbf[(size_t)(h * 128 + d) * 2048 + s] = f2bf(vc);
}

// ---------------------------------------------------------------------------
// Flash attention: one wave (32 threads) per (head, 16-query tile).
// Scores S = q k^T via WMMA (K = hs = 128 -> 4 chunks of 32), online softmax
// with bias = causal + -log1p(decay*log1p(dist)), PV via WMMA (K = 32 keys).
// P is re-fragmented C-layout -> A-layout through 1KB LDS.
// ---------------------------------------------------------------------------
__global__ __launch_bounds__(32) void attn_kernel(
    const unsigned short* __restrict__ qbf,   // [16][1024][128]
    const unsigned short* __restrict__ kbf,   // [16][2048][128]
    const unsigned short* __restrict__ vtbf,  // [16][128][2048]
    const float* __restrict__ decay_raw,      // [16]
    unsigned short* __restrict__ ybf) {       // [1024][2048]
  const int qt = blockIdx.x;   // 0..63
  const int h  = blockIdx.y;   // 0..15
  const int lane = threadIdx.x;
  const int lr = lane & 15, lh = lane >> 4;
  const int qi0 = qt * 16;

  __shared__ __align__(16) unsigned short Plds[16 * 32];

  const float decay = log1pf(__expf(decay_raw[h]));
  const float scale = 0.08838834764831845f;   // 1/sqrt(128)

  // q A-fragments (rows = qi0+lr), 4 chunks of K=32 over hs=128
  v16bf aq[4];
  {
    const unsigned short* qrow = qbf + (size_t)(h * 1024 + qi0 + lr) * 128;
#pragma unroll
    for (int c = 0; c < 4; ++c) aq[c] = loadAfrag(qrow + 32 * c, lh);
  }

  v8f acc[8];
  v8f z = {};
#pragma unroll
  for (int nt = 0; nt < 8; ++nt) acc[nt] = z;
  float m_run[8], l_run[8];
#pragma unroll
  for (int r = 0; r < 8; ++r) { m_run[r] = -3.0e38f; l_run[r] = 0.0f; }

  // keys needed: 0 .. (1024 + qi0 + 15); blocks of 32
  const int nblocks = (1024 + qi0 + 16 + 31) >> 5;
  for (int kb = 0; kb < nblocks; ++kb) {
    const int kb0 = kb * 32;

    // ---- scores: two 16x16 tiles covering keys kb0..kb0+31 ----
    v8f s[2];
#pragma unroll
    for (int t2 = 0; t2 < 2; ++t2) {
      const int key = kb0 + t2 * 16 + lr;          // B-frag: N = key
      const unsigned short* krow = kbf + (size_t)(h * 2048 + key) * 128 + lh * 16;
      v8f st = z;
#pragma unroll
      for (int c = 0; c < 4; ++c) {
        v16bf bk = load16(krow + 32 * c);          // K = 32c + lh*16 + e
        st = __builtin_amdgcn_wmma_f32_16x16x32_bf16(
            false, aq[c], false, bk, (short)0, st, false, false);
      }
      s[t2] = st;
    }

    // ---- bias + online softmax (row M = r + 8*lh, col = lr) ----
    float p0[8], p1[8], alpha[8];
#pragma unroll
    for (int r = 0; r < 8; ++r) {
      const int qpos = 1024 + qi0 + r + 8 * lh;
      const int k0i = kb0 + lr, k1i = kb0 + 16 + lr;
      float s0 = (k0i > qpos) ? -1e30f
               : s[0][r] * scale - log1pf(decay * log1pf((float)(qpos - k0i)));
      float s1 = (k1i > qpos) ? -1e30f
               : s[1][r] * scale - log1pf(decay * log1pf((float)(qpos - k1i)));
      float vmax = fmaxf(s0, s1);
      vmax = fmaxf(vmax, __shfl_xor(vmax, 1));
      vmax = fmaxf(vmax, __shfl_xor(vmax, 2));
      vmax = fmaxf(vmax, __shfl_xor(vmax, 4));
      vmax = fmaxf(vmax, __shfl_xor(vmax, 8));
      float mnew = fmaxf(m_run[r], vmax);
      float a  = __expf(m_run[r] - mnew);
      float e0 = __expf(s0 - mnew);
      float e1 = __expf(s1 - mnew);
      float rs = e0 + e1;
      rs += __shfl_xor(rs, 1);
      rs += __shfl_xor(rs, 2);
      rs += __shfl_xor(rs, 4);
      rs += __shfl_xor(rs, 8);
      l_run[r] = l_run[r] * a + rs;
      m_run[r] = mnew;
      alpha[r] = a;
      p0[r] = e0;
      p1[r] = e1;
    }
#pragma unroll
    for (int nt = 0; nt < 8; ++nt)
#pragma unroll
      for (int r = 0; r < 8; ++r) acc[nt][r] *= alpha[r];

    // ---- P: C-layout -> LDS [row][key] -> A-layout fragment ----
    __syncthreads();  // single-wave WG: protects LDS reuse vs prior reads
#pragma unroll
    for (int r = 0; r < 8; ++r) {
      const int row = r + 8 * lh;
      Plds[row * 32 + lr]      = f2bf(p0[r]);
      Plds[row * 32 + 16 + lr] = f2bf(p1[r]);
    }
    __syncthreads();
    v16bf ap = loadAfrag(&Plds[lr * 32], lh);

    // ---- PV: K = 32 keys, 8 feature tiles of 16 ----
    const unsigned short* vbase =
        vtbf + (size_t)(h * 128) * 2048 + kb0 + lh * 16;
#pragma unroll
    for (int nt = 0; nt < 8; ++nt) {
      v16bf bv = load16(vbase + (size_t)(nt * 16 + lr) * 2048);
      acc[nt] = __builtin_amdgcn_wmma_f32_16x16x32_bf16(
          false, ap, false, bv, (short)0, acc[nt], false, false);
    }
  }

  // ---- epilogue: y_bf[t][h*128 + feat] = acc / l ----
#pragma unroll
  for (int r = 0; r < 8; ++r) {
    const float inv = 1.0f / l_run[r];
    const int t = qi0 + r + 8 * lh;
#pragma unroll
    for (int nt = 0; nt < 8; ++nt)
      ybf[(size_t)t * 2048 + h * 128 + nt * 16 + lr] = f2bf(acc[nt][r] * inv);
  }
}

// ---------------------------------------------------------------------------
// Launch
// ---------------------------------------------------------------------------
extern "C" void kernel_launch(void* const* d_in, const int* in_sizes, int n_in,
                              void* d_out, int out_size, void* d_ws, size_t ws_size,
                              hipStream_t stream) {
  (void)in_sizes; (void)n_in; (void)out_size; (void)ws_size;

  const float* x         = (const float*)d_in[0];  // [1,1024,2048]
  const float* cached_k  = (const float*)d_in[1];  // [1,16,1024,128]
  const float* cached_v  = (const float*)d_in[2];  // [1,16,1024,128]
  const float* W_attn    = (const float*)d_in[3];  // [2048,6144]
  const float* b_attn    = (const float*)d_in[4];  // [6144]
  const float* W_proj    = (const float*)d_in[5];  // [2048,2048]
  const float* b_proj    = (const float*)d_in[6];  // [2048]
  const float* decay_raw = (const float*)d_in[7];  // [16]

  // d_out: y (2,097,152) | k (4,194,304) | v (4,194,304) fp32
  float* y    = (float*)d_out;
  float* kout = y + 2097152;
  float* vout = kout + 4194304;

  // workspace layout (~84 MB)
  char* ws = (char*)d_ws;
  const size_t OFF_XBF  = 0;                        // x bf16        (2,097,152)
  const size_t OFF_WA   = OFF_XBF + 2097152ull * 2; // W_attn^T bf16 (12,582,912)
  const size_t OFF_WP   = OFF_WA + 12582912ull * 2; // W_proj^T bf16 (4,194,304)
  const size_t OFF_QKV  = OFF_WP + 4194304ull * 2;  // qkv f32       (6,291,456)
  const size_t OFF_QBF  = OFF_QKV + 6291456ull * 4; // q bf16        (2,097,152)
  const size_t OFF_KBF  = OFF_QBF + 2097152ull * 2; // k bf16        (4,194,304)
  const size_t OFF_VTBF = OFF_KBF + 4194304ull * 2; // v^T bf16      (4,194,304)
  const size_t OFF_YBF  = OFF_VTBF + 4194304ull * 2;// y bf16        (2,097,152)

  unsigned short* xbf  = (unsigned short*)(ws + OFF_XBF);
  unsigned short* wabf = (unsigned short*)(ws + OFF_WA);   // [6144][2048]
  unsigned short* wpbf = (unsigned short*)(ws + OFF_WP);   // [2048][2048]
  float*          qkv  = (float*)(ws + OFF_QKV);
  unsigned short* qbf  = (unsigned short*)(ws + OFF_QBF);
  unsigned short* kbf  = (unsigned short*)(ws + OFF_KBF);
  unsigned short* vtbf = (unsigned short*)(ws + OFF_VTBF);
  unsigned short* ybf  = (unsigned short*)(ws + OFF_YBF);

  // 1) fp32 -> bf16 conversions (weights transposed to [N][K])
  cvt_kernel<<<8192, 256, 0, stream>>>(x, xbf, 2097152);
  cvt_transpose_kernel<<<49152, 256, 0, stream>>>(W_attn, wabf, 2048, 6144);
  cvt_transpose_kernel<<<16384, 256, 0, stream>>>(W_proj, wpbf, 2048, 2048);

  // 2) QKV projection: [1024,2048] x [2048,6144] + b_attn
  gemm_bf16_kernel<<<dim3(48, 8), 256, 0, stream>>>(xbf, wabf, b_attn, qkv,
                                                    1024, 6144, 2048);

  // 3) RoPE + KV concat (new tokens), cache copy (old tokens)
  rope_concat_kernel<<<1024, 256, 0, stream>>>(qkv, qbf, kbf, vtbf, kout, vout);
  cache_concat_kernel<<<8192, 256, 0, stream>>>(cached_k, cached_v, kbf, vtbf,
                                                kout, vout);

  // 4) flash attention: grid = (q tiles, heads), 1 wave per block
  attn_kernel<<<dim3(64, 16), 32, 0, stream>>>(qbf, kbf, vtbf, decay_raw, ybf);

  // 5) output projection: [1024,2048] x [2048,2048] + b_proj -> y
  gemm_bf16_kernel<<<dim3(16, 8), 256, 0, stream>>>(ybf, wpbf, b_proj, y,
                                                    1024, 2048, 2048);
}
